// multiFocusConvAttention_45913200394583
// MI455X (gfx1250) — matrix operand order repaired and checked by reading, not compiled, not run on hardware
//
#include <hip/hip_runtime.h>

typedef unsigned int u32;
typedef unsigned short u16;
typedef __attribute__((ext_vector_type(16))) _Float16 v16h;
typedef __attribute__((ext_vector_type(8)))  float    v8f;
typedef __attribute__((ext_vector_type(8)))  unsigned int v8u;

__device__ __forceinline__ u32 pack2(float a, float b) {
    unsigned short ha = __builtin_bit_cast(unsigned short, (_Float16)a);
    unsigned short hb = __builtin_bit_cast(unsigned short, (_Float16)b);
    return (u32)ha | ((u32)hb << 16);
}

__device__ __forceinline__ v16h to_v16h(uint4 lo, uint4 hi) {
    v8u u;
    u[0]=lo.x; u[1]=lo.y; u[2]=lo.z; u[3]=lo.w;
    u[4]=hi.x; u[5]=hi.y; u[6]=hi.z; u[7]=hi.w;
    return __builtin_bit_cast(v16h, u);
}

// async copy of 16 bytes: global -> LDS (per active lane), tracked on ASYNCcnt
__device__ __forceinline__ void async_g2l_b128(u32 lds_off, const void* gptr) {
    asm volatile("global_load_async_to_lds_b128 %0, %1, off"
                 :: "v"(lds_off), "v"((unsigned long long)(uintptr_t)gptr)
                 : "memory");
}
__device__ __forceinline__ void wait_async0() {
    asm volatile("s_wait_asynccnt 0x0" ::: "memory");
}

// -------------------------------------------------------------------------
// Kernel 1: pack weights into f16, per-WMMA-fragment layout.
// wkv: [ct(16)][ks(16)][lane(32)][v(8)] u32 (two f16)  -> 65536 u32
// wdr: [j(2)][ks(8)][lane(32)][v(8)] u32               -> 4096 u32
// K layout (kv): K = (kw*2+kh)*128 + c ; (dr): K = (kw*2+kh)*64 + cin
// fragment: N = tile*16 + (lane&15), g = lane>>4,
//           Klocal = (v&3)*2 + (v>>2)*16 + g*8
// -------------------------------------------------------------------------
__global__ void pack_weights(const float* __restrict__ key_w,
                             const float* __restrict__ value_w,
                             const float* __restrict__ dr_w,
                             u32* __restrict__ wkv, u32* __restrict__ wdr) {
    int id = blockIdx.x * 256 + threadIdx.x;
    if (id < 65536) {
        int v = id & 7, lane = (id >> 3) & 31, ks = (id >> 8) & 15, ct = id >> 12;
        int g = lane >> 4;
        int N = ct * 16 + (lane & 15);
        int K = ks * 32 + (v & 3) * 2 + (v >> 2) * 16 + g * 8;
        int kpos = K >> 7, c = K & 127, kw = kpos >> 1, kh = kpos & 1;
        float w0, w1;
        if (N < 128) {
            int head = N >> 4, ch = N & 15;
            const float* W = key_w + ((((head*2+kw)*2+kh)*128 + c) * 16 + ch);
            w0 = W[0]; w1 = W[16];
        } else {
            int head = (N - 128) >> 4, ch = N & 15;
            const float* W = value_w + ((((head*2+kw)*2+kh)*128 + c) * 16 + ch);
            w0 = W[0]; w1 = W[16];
        }
        wkv[id] = pack2(w0, w1);
    } else {
        int t = id - 65536;
        if (t < 4096) {
            int v = t & 7, lane = (t >> 3) & 31, ks = (t >> 8) & 7, j = t >> 11;
            int g = lane >> 4;
            int N = lane & 15;
            int K = ks * 32 + (v & 3) * 2 + (v >> 2) * 16 + g * 8;
            int kpos = K >> 6, cin = K & 63, kw = kpos >> 1, kh = kpos & 1;
            const float* W = dr_w + ((((j*2+kw)*2+kh)*64 + cin) * 16 + N);
            wdr[t] = pack2(W[0], W[16]);
        }
    }
}

// -------------------------------------------------------------------------
// Kernel 2: fused key+value conv for all 8 heads as one WMMA GEMM.
// M = 589,824 pixels (16 per WG), K = 512, N = 256. 8 waves per WG, each
// wave owns 2 N-tiles (waves 0-3 -> key cols, waves 4-7 -> value cols).
// The full fp32 x halo tile (2 x 17 x 128, rows padded to 132 floats
// against bank conflicts) is staged in LDS ONCE via
// GLOBAL_LOAD_ASYNC_TO_LDS_B128; border zero-filled via ds_store. The
// 16-step K-loop is then barrier-free: ds_load_b128 + cvt f32->f16 + WMMA.
// -------------------------------------------------------------------------
#define KV_ROW 132   // 128 floats + 4 pad  (132*4 B row stride; 132%64==4)
__global__ __launch_bounds__(256) void kv_gemm(const float* __restrict__ x,
                                               const float* __restrict__ key_b,
                                               const float* __restrict__ value_b,
                                               const u32* __restrict__ wkv,
                                               float* __restrict__ kbuf,
                                               float* __restrict__ vbuf) {
    __shared__ __align__(16) float xs[2 * 17 * KV_ROW];   // 17,952 B
    int tile = blockIdx.x;                      // 64*96*6 = 36864 tiles
    int bt = tile / 576;
    int rem = tile % 576;
    int w  = rem / 6;
    int h0 = (rem % 6) * 16;
    int tid = threadIdx.x, lane = tid & 31, wv = tid >> 5;
    int laneN = lane & 15, g = lane >> 4;

    // ---- stage x halo: 2 planes x 17 rows x 32 uint4 = 1088 transfers ----
    u32 ldsbase = (u32)(uintptr_t)xs;
    for (int it = 0; it < 5; ++it) {
        int idx = tid + it * 256;
        if (idx < 1088) {
            int kwp = idx / 544;
            int rr  = (idx - kwp * 544) >> 5;   // 0..16
            int ch  = idx & 31;                 // uint4 within row
            int hh = h0 + rr;
            int loff = ((kwp * 17 + rr) * KV_ROW + ch * 4);
            if ((w + kwp) < 96 && hh < 96) {
                const float* gp = x + (size_t)((bt*96 + w + kwp)*96 + hh) * 128 + ch * 4;
                async_g2l_b128(ldsbase + (u32)(loff * 4), gp);
            } else {
                *(uint4*)(xs + loff) = make_uint4(0u, 0u, 0u, 0u);
            }
        }
    }
    wait_async0();
    __syncthreads();

    // ---- accumulators init = bias (bias depends only on N column) ----
    int col0 = wv * 32 + laneN;                 // wave-uniform k/v split
    int col1 = col0 + 16;
    const float* bsrc = (wv < 4) ? key_b : value_b;
    float b0 = bsrc[col0 & 127];
    float b1 = bsrc[col1 & 127];
    v8f acc0, acc1;
#pragma unroll
    for (int r = 0; r < 8; ++r) { acc0[r] = b0; acc1[r] = b1; }

    const uint4* bbase = (const uint4*)wkv;
    for (int ks = 0; ks < 16; ++ks) {
        int kpos = ks >> 2, kw = kpos >> 1, kh = kpos & 1, c0 = (ks & 3) * 32;

        // A fragment: 16 fp32 from LDS, convert+pack to f16 pairs
        const float* ap = xs + (kw * 17 + laneN + kh) * KV_ROW + c0 + g * 8;
        float4 A0 = *(const float4*)(ap);
        float4 A1 = *(const float4*)(ap + 4);
        float4 A2 = *(const float4*)(ap + 16);
        float4 A3 = *(const float4*)(ap + 20);
        v8u a8;
        a8[0] = pack2(A0.x, A0.y); a8[1] = pack2(A0.z, A0.w);
        a8[2] = pack2(A1.x, A1.y); a8[3] = pack2(A1.z, A1.w);
        a8[4] = pack2(A2.x, A2.y); a8[5] = pack2(A2.z, A2.w);
        a8[6] = pack2(A3.x, A3.y); a8[7] = pack2(A3.z, A3.w);
        v16h av = __builtin_bit_cast(v16h, a8);

        const uint4* bp = bbase + (((wv*2)*16 + ks) * 64 + lane * 2);
        uint4 b0lo = bp[0],     b0hi = bp[1];
        uint4 b1lo = bp[1024],  b1hi = bp[1025];  // next N-tile: +16*256/4 uint4
        v16h bv0 = to_v16h(b0lo, b0hi);
        v16h bv1 = to_v16h(b1lo, b1hi);

        acc0 = __builtin_amdgcn_wmma_f32_16x16x32_f16(false, av, false, bv0,
                                                      (short)0, acc0, false, false);
        acc1 = __builtin_amdgcn_wmma_f32_16x16x32_f16(false, av, false, bv1,
                                                      (short)0, acc1, false, false);
    }

    // ---- epilogue: wave-uniform destination, no per-lane divergence ----
    float* obase = (wv < 4) ? kbuf : vbuf;
    int cc0 = col0 & 127, cc1 = col1 & 127;
    int pixbase = (bt*96 + w)*96 + h0;
#pragma unroll
    for (int r = 0; r < 8; ++r) {
        size_t row = (size_t)(pixbase + r + g * 8) * 128;
        obase[row + cc0] = acc0[r];
        obase[row + cc1] = acc1[r];
    }
}

// -------------------------------------------------------------------------
// Kernel 3: key_att = conv2x2(k, att_w[:, :, :16, :]) + att_b  (K=64, VALU)
// kabuf layout: [pix][head]
// -------------------------------------------------------------------------
__global__ __launch_bounds__(256) void ka_kernel(const float* __restrict__ kbuf,
                                                 const float* __restrict__ att_w,
                                                 const float* __restrict__ att_b,
                                                 float* __restrict__ kabuf) {
    int id = blockIdx.x * 256 + threadIdx.x;    // 4,718,592
    int head = id & 7;
    int pix = id >> 3;
    int h = pix % 96, w = (pix / 96) % 96, bt = pix / 9216;
    float acc = att_b[head];
#pragma unroll
    for (int kw = 0; kw < 2; ++kw)
#pragma unroll
        for (int kh = 0; kh < 2; ++kh) {
            int ww = w + kw, hh = h + kh;
            if (ww < 96 && hh < 96) {
                const float* kp = kbuf + (size_t)((bt*96 + ww)*96 + hh)*128 + head*16;
                const float* aw = att_w + ((head*2 + kw)*2 + kh) * 32;
#pragma unroll
                for (int c = 0; c < 16; ++c) acc += kp[c] * aw[c];
            }
        }
    kabuf[(size_t)pix * 8 + head] = acc;
}

// -------------------------------------------------------------------------
// Kernel 4: per (j,i,b,w,h): ta from selector parity, relu, softmax over T,
// weighted sum of v over T for all 4 (dx,dy) variants; write concat4 (f16).
// concat4 layout: [j*8+i][b][w][h][var*16+c]  (feeds dr_gemm A-matrix)
// -------------------------------------------------------------------------
__global__ __launch_bounds__(256) void attn_kernel(const float* __restrict__ vbuf,
                                                   const float* __restrict__ kabuf,
                                                   const float* __restrict__ selectors,
                                                   const float* __restrict__ att_w,
                                                   u32* __restrict__ concat4) {
    int id = blockIdx.x * 256 + threadIdx.x;    // 589,824
    int pixout = id % 36864;
    int rest = id / 36864;                      // j*8+i
    int j = rest >> 3, i = rest & 7;
    int h = pixout % 96, w = (pixout / 96) % 96, b = pixout / 9216;
    int pw = w * 96 + h;

    float ka[16];
#pragma unroll
    for (int t = 0; t < 16; ++t)
        ka[t] = kabuf[(size_t)((b*16 + t)*9216 + pw) * 8 + i];

    const float* aw = att_w + (i*4) * 32 + 16;  // att_w[i][kw][kh][16+c]
    const float* sj = selectors + j * 64;       // [2][2][16]

    float soft[4][16];
#pragma unroll
    for (int var = 0; var < 4; ++var) {
        int dx = var >> 1, dy = var & 1;
        float ta = 0.f;
#pragma unroll
        for (int kw = 0; kw < 2; ++kw)
#pragma unroll
            for (int kh = 0; kh < 2; ++kh) {
                if ((w + kw) < 96 && (h + kh) < 96) {
                    const float* sp = sj + ((((w+kw+dx)&1)*2 + ((h+kh+dy)&1)) * 16);
                    const float* apt = aw + (kw*2 + kh) * 32;
#pragma unroll
                    for (int c = 0; c < 16; ++c) ta += sp[c] * apt[c];
                }
            }
        float a[16], m = 0.f;
#pragma unroll
        for (int t = 0; t < 16; ++t) { a[t] = fmaxf(ka[t] + ta, 0.f); m = fmaxf(m, a[t]); }
        float s = 0.f;
#pragma unroll
        for (int t = 0; t < 16; ++t) { a[t] = __expf(a[t] - m); s += a[t]; }
        float inv = 1.f / s;
#pragma unroll
        for (int t = 0; t < 16; ++t) soft[var][t] = a[t] * inv;
    }

    float outv[4][16];
#pragma unroll
    for (int var = 0; var < 4; ++var)
#pragma unroll
        for (int c = 0; c < 16; ++c) outv[var][c] = 0.f;

    for (int t = 0; t < 16; ++t) {
        const float* vp = vbuf + (size_t)((b*16 + t)*9216 + pw) * 128 + i * 16;
        float vv[16];
#pragma unroll
        for (int c = 0; c < 16; ++c) vv[c] = vp[c];
#pragma unroll
        for (int var = 0; var < 4; ++var) {
            float sw = soft[var][t];
#pragma unroll
            for (int c = 0; c < 16; ++c) outv[var][c] += vv[c] * sw;
        }
    }

    u32* dst = concat4 + ((size_t)rest * 36864 + pixout) * 32;  // 64 f16 = 32 u32
#pragma unroll
    for (int var = 0; var < 4; ++var)
#pragma unroll
        for (int c = 0; c < 8; ++c)
            dst[var*8 + c] = pack2(outv[var][2*c], outv[var][2*c + 1]);
}

// -------------------------------------------------------------------------
// Kernel 5: dr conv as WMMA GEMM per (j,i) pair: M-tile 16 pixels, K=256
// (8 K-steps), N=16; one wave32 per workgroup. A halo (2 x 17 x 64 f16,
// rows padded to 72 halves) staged in LDS via async copy, border zeroed,
// then a branch-free K-loop of ds_load_b128 + WMMA. Fused +bias, ReLU.
// -------------------------------------------------------------------------
#define DR_ROW 72    // 64 halves + 8 pad (144 B row stride; 36%64 spreads banks)
__global__ __launch_bounds__(32) void dr_gemm(const u32* __restrict__ concat4,
                                              const u32* __restrict__ wdr,
                                              const float* __restrict__ dr_b,
                                              float* __restrict__ out) {
    __shared__ __align__(16) u16 atile[2 * 17 * DR_ROW];   // 4,896 B
    int wg = blockIdx.x;                        // 16 * 2304 = 36864
    int pair = wg / 2304;                       // j*8+i
    int tile = wg % 2304;
    int j = pair >> 3;
    int b = tile / 576;
    int rem = tile % 576;
    int w  = rem / 6;
    int h0 = (rem % 6) * 16;
    int lane = threadIdx.x & 31, laneN = lane & 15, g = lane >> 4;

    // ---- stage A halo: 2 planes x 17 rows x 8 uint4 = 272 transfers ----
    u32 ldsbase = (u32)(uintptr_t)atile;
    const u16* csrc = (const u16*)concat4;
    for (int it = 0; it < 9; ++it) {
        int idx = lane + it * 32;
        if (idx < 272) {
            int kwp = idx / 136;
            int rr  = (idx - kwp * 136) >> 3;   // 0..16
            int ch  = idx & 7;                  // uint4 within row
            int hh = h0 + rr;
            int loff = (kwp * 17 + rr) * DR_ROW + ch * 8;   // in halves
            if ((w + kwp) < 96 && hh < 96) {
                const u16* gp = csrc +
                    ((size_t)pair * 36864 + (b*9216 + (w + kwp)*96 + hh)) * 64 + ch * 8;
                async_g2l_b128(ldsbase + (u32)(loff * 2), gp);
            } else {
                *(uint4*)(atile + loff) = make_uint4(0u, 0u, 0u, 0u);
            }
        }
    }
    wait_async0();   // single wave: LDS same-wave ordering covers the ds stores

    v8f acc;
    float bias = dr_b[j * 16 + laneN];
#pragma unroll
    for (int r = 0; r < 8; ++r) acc[r] = bias;

    const uint4* bbase = (const uint4*)wdr;
    for (int ks = 0; ks < 8; ++ks) {
        int kpos = ks >> 1, kw = kpos >> 1, kh = kpos & 1, c0 = (ks & 1) * 32;

        const u16* ap = atile + (kw * 17 + laneN + kh) * DR_ROW + c0 + g * 8;
        uint4 alo = *(const uint4*)(ap);
        uint4 ahi = *(const uint4*)(ap + 16);
        v16h av = to_v16h(alo, ahi);

        const uint4* bp = bbase + (((j*8 + ks) * 32 + lane) * 2);
        v16h bv = to_v16h(bp[0], bp[1]);

        acc = __builtin_amdgcn_wmma_f32_16x16x32_f16(false, av, false, bv,
                                                     (short)0, acc, false, false);
    }

    int pixbase = (b*96 + w)*96 + h0;
#pragma unroll
    for (int r = 0; r < 8; ++r) {
        int pix = pixbase + r + g * 8;
        out[(size_t)pix * 256 + pair * 16 + laneN] = fmaxf(acc[r], 0.f);
    }
}

// -------------------------------------------------------------------------
// Host-side launcher
// -------------------------------------------------------------------------
extern "C" void kernel_launch(void* const* d_in, const int* in_sizes, int n_in,
                              void* d_out, int out_size, void* d_ws, size_t ws_size,
                              hipStream_t stream) {
    const float* x         = (const float*)d_in[0];
    const float* selectors = (const float*)d_in[1];
    const float* key_w     = (const float*)d_in[2];
    const float* key_b     = (const float*)d_in[3];
    const float* value_w   = (const float*)d_in[4];
    const float* value_b   = (const float*)d_in[5];
    const float* att_w     = (const float*)d_in[6];
    const float* att_b     = (const float*)d_in[7];
    const float* dr_w      = (const float*)d_in[8];
    const float* dr_b      = (const float*)d_in[9];
    float* out = (float*)d_out;

    char* ws = (char*)d_ws;
    const size_t KV_BYTES = 589824ull * 128ull * 4ull;   // 301,989,888 each
    u32*   wkv    = (u32*)(ws);                          // 262,144 B
    u32*   wdr    = (u32*)(ws + 262144);                 // 16,384 B
    float* kbuf   = (float*)(ws + 278528);
    float* vbuf   = (float*)(ws + 278528 + KV_BYTES);
    float* kabuf  = (float*)(ws + 278528 + 2*KV_BYTES);  // 18,874,368 B
    u32*   concat4 = (u32*)kbuf;   // alias: kbuf is dead after ka_kernel

    pack_weights<<<272, 256, 0, stream>>>(key_w, value_w, dr_w, wkv, wdr);
    kv_gemm<<<36864, 256, 0, stream>>>(x, key_b, value_b, wkv, kbuf, vbuf);
    ka_kernel<<<18432, 256, 0, stream>>>(kbuf, att_w, att_b, kabuf);
    attn_kernel<<<2304, 256, 0, stream>>>(vbuf, kabuf, selectors, att_w, concat4);
    dr_gemm<<<36864, 32, 0, stream>>>(concat4, wdr, dr_b, out);
}
